// StochasticKeyNet_84945863180390
// MI455X (gfx1250) — compile-verified, matrix operand order checked by reading, not run on hardware
//
#include <hip/hip_runtime.h>

typedef __attribute__((ext_vector_type(16))) _Float16 v16h;
typedef __attribute__((ext_vector_type(8)))  _Float16 v8h;
typedef __attribute__((ext_vector_type(4)))  _Float16 v4h;
typedef __attribute__((ext_vector_type(8)))  float    v8f;

// ---------------------------------------------------------------------------
// Decode initial key into NHWC, channels padded 3->32 (pad pre-zeroed):
//   pi = c*1024 + h*32 + w  (NCHW flat)  ->  x[(b*1024 + hw)*32 + c]
// ---------------------------------------------------------------------------
__global__ void decode_kernel(const float* __restrict__ y, const int* __restrict__ p,
                              const float* __restrict__ s, _Float16* __restrict__ x,
                              int total) {
  int idx = blockIdx.x * blockDim.x + threadIdx.x;
  if (idx >= total) return;
  int b = idx / 3073;
  int i = idx - b * 3073;
  int pi = p[i];
  if (pi < 3072) {
    int c  = pi >> 10;
    int hw = pi & 1023;
    x[(b * 1024 + hw) * 32 + c] = (_Float16)(y[idx] / s[i]);
  }
}

// ---------------------------------------------------------------------------
// Weight prep -> f16 Wt[Npad][Kpad], NHWC K-order: k = (kh*R+kw)*CinPad + cin.
// Handles channel padding (cin >= CinReal -> 0) and N padding. Also covers
// fc2 (R=1, CinReal=100, CinPad=128).
// ---------------------------------------------------------------------------
__global__ void wprep_conv_kernel(const float* __restrict__ w, _Float16* __restrict__ wt,
                                  int Cout, int CinReal, int CinPad, int R,
                                  int Kpad, int total) {
  int idx = blockIdx.x * blockDim.x + threadIdx.x;
  if (idx >= total) return;
  int n = idx / Kpad;
  int k = idx - n * Kpad;
  int khw = k / CinPad;
  int cin = k - khw * CinPad;
  int kh = khw / R;
  int kw = khw - kh * R;
  float v = 0.0f;
  if (n < Cout && cin < CinReal) v = w[((n * CinReal + cin) * R + kh) * R + kw];
  wt[idx] = (_Float16)v;
}

// fc1: absorbs the NCHW flatten of conv9's output (j = c*64 + hw) into the
// NHWC-padded k-order (k = hw*16 + c, c padded 10->16). Kpad = 1024.
__global__ void wprep_fc1_kernel(const float* __restrict__ w, _Float16* __restrict__ wt,
                                 int total) {
  int idx = blockIdx.x * blockDim.x + threadIdx.x;
  if (idx >= total) return;
  int n = idx >> 10;          // Npad = 128 rows
  int k = idx & 1023;
  int c  = k & 15;
  int hw = k >> 4;
  float v = 0.0f;
  if (n < 100 && c < 10) v = w[n * 640 + (c * 64 + hw)];
  wt[idx] = (_Float16)v;
}

// ---------------------------------------------------------------------------
// Implicit-GEMM conv, NHWC, fully compile-time specialized.
//   A[m,k]: m=(b,oh,ow), k=(kh,kw,cin) -> contiguous 32-ch chunks (CIN%32==0)
//   B[k,n] = Wt[n][k]
// Block: 256 thr = 8 waves; tile MT x 32 (MT=256 conv / 128 fc); K step 32.
// Per wave: MSUB=MT/128 M-subtiles x 2 N-subtiles -> 2*MSUB wmma per K-step,
// B fragments reused across M-subtiles, A across N-subtiles.
// Double-buffered LDS with register-staged global prefetch.
// ---------------------------------------------------------------------------
template <int MT, int R, int CIN, int H, int W, int COUT, int STRIDE, int CSTORE>
__global__ __launch_bounds__(256) void conv_wmma_kernel(
    const _Float16* __restrict__ X, const _Float16* __restrict__ Wt,
    const float* __restrict__ bias, _Float16* __restrict__ Yh,
    float* __restrict__ Yf, int doRelu) {
  constexpr int PAD  = (R - 1) / 2;
  constexpr int HO   = (H + 2 * PAD - R) / STRIDE + 1;
  constexpr int WO   = (W + 2 * PAD - R) / STRIDE + 1;
  constexpr int OHW  = HO * WO;                // power of two
  constexpr int K    = R * R * CIN;            // multiple of 32
  constexpr int MSUB = MT / 128;               // 1 (fc) or 2 (conv)

  __shared__ __align__(16) _Float16 Alds[2][MT][40];
  __shared__ __align__(16) _Float16 Blds[2][32][40];

  const int tid  = threadIdx.x;
  const int wave = tid >> 5;
  const int lane = tid & 31;
  const int mBase = blockIdx.x * MT;
  const int nBase = blockIdx.y * 32;

  // A staging: each thread stages 16 contiguous halves per 128-row sub-tile
  const int arow = tid >> 1;
  const int akk0 = (tid & 1) << 4;
  int ab[MSUB], aoh[MSUB], aow[MSUB];
#pragma unroll
  for (int s = 0; s < MSUB; ++s) {
    int am   = mBase + s * 128 + arow;
    ab[s]    = am / OHW;                       // shifts (OHW = 2^n)
    int arem = am - ab[s] * OHW;
    aoh[s]   = arem / WO;
    aow[s]   = arem - aoh[s] * WO;
  }

  // B staging: 4 halves per thread
  const int brow = tid >> 3;
  const int bkk  = (tid & 7) << 2;

  v8f acc[MSUB][2];
#pragma unroll
  for (int s = 0; s < MSUB; ++s) {
    v8f z = {};
    acc[s][0] = z;
    acc[s][1] = z;
  }

  auto loadA = [&](int kt, int s, v8h& r0, v8h& r1) {
    const int k0   = kt + akk0;
    const int khw  = k0 / CIN;                 // compile-time divisor
    const int cin0 = k0 - khw * CIN;
    int kh, kw;
    if (R == 3) { kh = khw / 3; kw = khw - kh * 3; } else { kh = 0; kw = 0; }
    const int ih = aoh[s] * STRIDE + kh - PAD;
    const int iw = aow[s] * STRIDE + kw - PAD;
    v8h z = {};
    r0 = z; r1 = z;
    if (ih >= 0 && ih < H && iw >= 0 && iw < W) {
      const v8h* src =
          (const v8h*)(X + ((size_t)((ab[s] * H + ih) * W + iw) * CIN + cin0));
      r0 = src[0];
      r1 = src[1];
    }
  };
  auto loadB = [&](int kt, v4h& rbv) {
    rbv = *(const v4h*)(Wt + (size_t)(nBase + brow) * K + kt + bkk);
  };

  // prologue: stage tile 0
  v8h ra[MSUB][2];
  v4h rb;
#pragma unroll
  for (int s = 0; s < MSUB; ++s) loadA(0, s, ra[s][0], ra[s][1]);
  loadB(0, rb);
#pragma unroll
  for (int s = 0; s < MSUB; ++s) {
    *(v8h*)&Alds[0][s * 128 + arow][akk0]     = ra[s][0];
    *(v8h*)&Alds[0][s * 128 + arow][akk0 + 8] = ra[s][1];
  }
  *(v4h*)&Blds[0][brow][bkk] = rb;

  const int rBase = wave * (16 * MSUB) + (lane & 15);
  const int kh8   = (lane >> 4) << 3;
  const int n0    = lane & 15;

#pragma unroll 2
  for (int kt = 0, buf = 0; kt < K; kt += 32, buf ^= 1) {
    __syncthreads();
    const bool more = (kt + 32) < K;
    if (more) {                                // prefetch next tile into regs
#pragma unroll
      for (int s = 0; s < MSUB; ++s) loadA(kt + 32, s, ra[s][0], ra[s][1]);
      loadB(kt + 32, rb);
    }
    // fragments per CDNA5 16-bit WMMA layout:
    // lanes 0-15 -> row, K {0..7,16..23}; lanes 16-31 -> row, K {8..15,24..31}
    v16h bfrag[2];
    {
      v8h x0 = *(const v8h*)&Blds[buf][n0][kh8];
      v8h x1 = *(const v8h*)&Blds[buf][n0][16 + kh8];
      for (int j = 0; j < 8; ++j) { bfrag[0][j] = x0[j]; bfrag[0][8 + j] = x1[j]; }
      v8h y0 = *(const v8h*)&Blds[buf][16 + n0][kh8];
      v8h y1 = *(const v8h*)&Blds[buf][16 + n0][16 + kh8];
      for (int j = 0; j < 8; ++j) { bfrag[1][j] = y0[j]; bfrag[1][8 + j] = y1[j]; }
    }
#pragma unroll
    for (int s = 0; s < MSUB; ++s) {
      v16h a;
      v8h x0 = *(const v8h*)&Alds[buf][rBase + s * 16][kh8];
      v8h x1 = *(const v8h*)&Alds[buf][rBase + s * 16][16 + kh8];
      for (int j = 0; j < 8; ++j) { a[j] = x0[j]; a[8 + j] = x1[j]; }
      acc[s][0] = __builtin_amdgcn_wmma_f32_16x16x32_f16(
          false, a, false, bfrag[0], (short)0, acc[s][0], false, false);
      acc[s][1] = __builtin_amdgcn_wmma_f32_16x16x32_f16(
          false, a, false, bfrag[1], (short)0, acc[s][1], false, false);
    }
    if (more) {                                // fill the other buffer
#pragma unroll
      for (int s = 0; s < MSUB; ++s) {
        *(v8h*)&Alds[buf ^ 1][s * 128 + arow][akk0]     = ra[s][0];
        *(v8h*)&Alds[buf ^ 1][s * 128 + arow][akk0 + 8] = ra[s][1];
      }
      *(v4h*)&Blds[buf ^ 1][brow][bkk] = rb;
    }
  }

  // epilogue: D layout: VGPR v -> M = v + 8*(lane>=16), N = lane&15
  const int nCol0 = nBase + (lane & 15);
  const int nCol1 = nCol0 + 16;
#pragma unroll
  for (int s = 0; s < MSUB; ++s) {
    const int mRowBase = wave * (16 * MSUB) + s * 16 + ((lane >> 4) << 3);
    for (int v = 0; v < 8; ++v) {
      int m   = mBase + mRowBase + v;
      int b   = m / OHW;
      int rem = m - b * OHW;
      int oh  = rem / WO;
      int ow  = rem - oh * WO;
      size_t pix = (size_t)((b * HO + oh) * WO + ow) * CSTORE;
      if (nCol0 < COUT) {
        float vv = acc[s][0][v] + bias[nCol0];
        if (doRelu) vv = fmaxf(vv, 0.0f);
        if (Yh) Yh[pix + nCol0] = (_Float16)vv; else Yf[pix + nCol0] = vv;
      }
      if (nCol1 < COUT) {
        float vv = acc[s][1][v] + bias[nCol1];
        if (doRelu) vv = fmaxf(vv, 0.0f);
        if (Yh) Yh[pix + nCol1] = (_Float16)vv; else Yf[pix + nCol1] = vv;
      }
    }
  }
}

// ---------------------------------------------------------------------------
// Final encode: out[b,j] = s11[j] * concat(fc2[b,:], 1)[p11[j]]
// ---------------------------------------------------------------------------
__global__ void encode_kernel(const float* __restrict__ fc2, const int* __restrict__ p,
                              const float* __restrict__ s, float* __restrict__ out,
                              int total) {
  int idx = blockIdx.x * blockDim.x + threadIdx.x;
  if (idx >= total) return;
  int b = idx / 11;
  int j = idx - b * 11;
  int pj = p[j];
  float v = (pj < 10) ? fc2[b * 10 + pj] : 1.0f;
  out[idx] = s[j] * v;
}

// ---------------------------------------------------------------------------
static void prep_conv(hipStream_t st, const float* w, _Float16* wt, int Cout,
                      int CinReal, int CinPad, int R) {
  int Kpad  = R * R * CinPad;
  int Npad  = (Cout + 31) & ~31;
  int total = Npad * Kpad;
  wprep_conv_kernel<<<(total + 255) / 256, 256, 0, st>>>(w, wt, Cout, CinReal,
                                                         CinPad, R, Kpad, total);
}

template <int MT, int R, int CIN, int H, int W, int COUT, int STRIDE, int CSTORE>
static void launch_conv(hipStream_t st, const _Float16* X, const _Float16* Wt,
                        const float* bias, _Float16* Yh, float* Yf, int doRelu) {
  constexpr int PAD  = (R - 1) / 2;
  constexpr int HO   = (H + 2 * PAD - R) / STRIDE + 1;
  constexpr int WO   = (W + 2 * PAD - R) / STRIDE + 1;
  constexpr int M    = 128 * HO * WO;
  constexpr int NPAD = (COUT + 31) & ~31;
  static_assert(M % MT == 0, "M tile mismatch");
  dim3 grid(M / MT, NPAD / 32);
  conv_wmma_kernel<MT, R, CIN, H, W, COUT, STRIDE, CSTORE>
      <<<grid, 256, 0, st>>>(X, Wt, bias, Yh, Yf, doRelu);
}

extern "C" void kernel_launch(void* const* d_in, const int* in_sizes, int n_in,
                              void* d_out, int out_size, void* d_ws, size_t ws_size,
                              hipStream_t stream) {
  (void)in_sizes; (void)n_in; (void)out_size; (void)ws_size;
  const int*   p0  = (const int*)d_in[0];
  const float* s0  = (const float*)d_in[1];
  const int*   p11 = (const int*)d_in[26];
  const float* s11 = (const float*)d_in[27];
  const float* Wc[9];
  const float* Bc[9];
  for (int i = 0; i < 9; ++i) {
    Wc[i] = (const float*)d_in[28 + 2 * i];
    Bc[i] = (const float*)d_in[29 + 2 * i];
  }
  const float* fw1 = (const float*)d_in[46];
  const float* fb1 = (const float*)d_in[47];
  const float* fw2 = (const float*)d_in[48];
  const float* fb2 = (const float*)d_in[49];
  const float* A0  = (const float*)d_in[50];

  char* ws = (char*)d_ws;
  _Float16* bufA   = (_Float16*)(ws);                        // <= 24 MB NHWC acts
  _Float16* bufB   = (_Float16*)(ws + ((size_t)25 << 20));   // <= 24 MB NHWC acts
  _Float16* wgt    = (_Float16*)(ws + ((size_t)50 << 20));   // <= 1 MB staged weights
  float*    fc2out = (float*)   (ws + ((size_t)51 << 20));   // 128*10 f32

  // 1) zero-padded NHWC input [128][32][32][32], then decode scatter
  hipMemsetAsync(bufA, 0, (size_t)128 * 1024 * 32 * sizeof(_Float16), stream);
  {
    int total = 128 * 3073;
    decode_kernel<<<(total + 255) / 256, 256, 0, stream>>>(A0, p0, s0, bufA, total);
  }

  // 2) CNN (inter-layer enc/dec keys cancel algebraically)
  prep_conv(stream, Wc[0], wgt, 96, 3, 32, 3);
  launch_conv<256, 3, 32, 32, 32, 96, 1, 96>(stream, bufA, wgt, Bc[0], bufB, nullptr, 1);
  prep_conv(stream, Wc[1], wgt, 96, 96, 96, 3);
  launch_conv<256, 3, 96, 32, 32, 96, 1, 96>(stream, bufB, wgt, Bc[1], bufA, nullptr, 1);
  prep_conv(stream, Wc[2], wgt, 96, 96, 96, 3);
  launch_conv<256, 3, 96, 32, 32, 96, 2, 96>(stream, bufA, wgt, Bc[2], bufB, nullptr, 1); // ->16x16
  prep_conv(stream, Wc[3], wgt, 192, 96, 96, 3);
  launch_conv<256, 3, 96, 16, 16, 192, 1, 192>(stream, bufB, wgt, Bc[3], bufA, nullptr, 1);
  prep_conv(stream, Wc[4], wgt, 192, 192, 192, 3);
  launch_conv<256, 3, 192, 16, 16, 192, 1, 192>(stream, bufA, wgt, Bc[4], bufB, nullptr, 1);
  prep_conv(stream, Wc[5], wgt, 192, 192, 192, 3);
  launch_conv<256, 3, 192, 16, 16, 192, 2, 192>(stream, bufB, wgt, Bc[5], bufA, nullptr, 1); // ->8x8
  prep_conv(stream, Wc[6], wgt, 192, 192, 192, 3);
  launch_conv<256, 3, 192, 8, 8, 192, 1, 192>(stream, bufA, wgt, Bc[6], bufB, nullptr, 1);
  prep_conv(stream, Wc[7], wgt, 192, 192, 192, 1);
  launch_conv<256, 1, 192, 8, 8, 192, 1, 192>(stream, bufB, wgt, Bc[7], bufA, nullptr, 1);   // 1x1
  // conv9: Cout 10 stored padded to 16 channels (pad pre-zeroed for fc1)
  hipMemsetAsync(bufB, 0, (size_t)128 * 64 * 16 * sizeof(_Float16), stream);
  prep_conv(stream, Wc[8], wgt, 10, 192, 192, 1);
  launch_conv<256, 1, 192, 8, 8, 10, 1, 16>(stream, bufA, wgt, Bc[8], bufB, nullptr, 1);     // 1x1

  // 3) fc1 as 1x1 conv over flat [128][1024]; weight prep absorbs NCHW flatten
  hipMemsetAsync(bufA, 0, (size_t)128 * 128 * sizeof(_Float16), stream);
  {
    int total = 128 * 1024;
    wprep_fc1_kernel<<<(total + 255) / 256, 256, 0, stream>>>(fw1, wgt, total);
  }
  launch_conv<128, 1, 1024, 1, 1, 100, 1, 128>(stream, bufB, wgt, fb1, bufA, nullptr, 1);

  // 4) fc2: K padded 100->128 (zeros in both act pad and weight pad) -> f32
  prep_conv(stream, fw2, wgt, 10, 100, 128, 1);
  launch_conv<128, 1, 128, 1, 1, 10, 1, 10>(stream, bufA, wgt, fb2, nullptr, fc2out, 0);

  // 5) final encode with (p11, s11)
  {
    int total = 128 * 11;
    encode_kernel<<<(total + 255) / 256, 256, 0, stream>>>(fc2out, p11, s11,
                                                           (float*)d_out, total);
  }
}
